// RNN_51505247813862
// MI455X (gfx1250) — matrix operand order repaired and checked by reading, not compiled, not run on hardware
//
#include <hip/hip_runtime.h>

// ---------------------------------------------------------------------------
// RNN on MI455X (gfx1250):
//   1) xin = x @ W_in^T   : parallel f32 WMMA (16x16x4) -> rnn_out temp region
//   2) recurrent scan     : persistent kernel, bf16x3 split-precision GEMM on
//                           v_wmma_f32_16x16x32_bf16, f32 state carry, grid
//                           step-barrier via agent-scope atomics
//   3) out = rnn @ W_out^T: parallel f32 WMMA (16x16x4)
// ---------------------------------------------------------------------------

typedef float v2f  __attribute__((ext_vector_type(2)));
typedef float v8f  __attribute__((ext_vector_type(8)));
typedef __bf16 bf16_t;
typedef bf16_t v8bf  __attribute__((ext_vector_type(8)));
typedef bf16_t v16bf __attribute__((ext_vector_type(16)));

#define T_STEPS 512
#define BATCH   64
#define IDIM    128
#define HDIM    512
#define ODIM    64
#define ALPHA_F       0.2f
#define NOISE_SCALE_F 0.0632455532f   // 0.1 * sqrt(2*0.2)

#define NWG 32        // scan workgroups; each owns CW hidden columns
#define CW  16        // hidden columns per workgroup (one 16-wide WMMA tile)

// Workspace: [0..255] sync counter; then hbHi (64KB), hbLo (64KB):
//   hb*[b*HDIM + k]  row-major bf16 hidden state (hi / lo split)
#define HB_HI_OFF 256
#define HB_LO_OFF (256 + BATCH * HDIM * 2)

static __device__ __forceinline__ v8f wmma_f32(v2f a, v2f b, v8f c) {
    return __builtin_amdgcn_wmma_f32_16x16x4_f32(
        false, a, false, b, (short)0, c, false, false);
}
static __device__ __forceinline__ v8f wmma_bf16(v16bf a, v16bf b, v8f c) {
    return __builtin_amdgcn_wmma_f32_16x16x32_bf16(
        false, a, false, b, (short)0, c, false, false);
}
static __device__ __forceinline__ v16bf cat16(v8bf a, v8bf b) {
    return __builtin_shufflevector(a, b, 0, 1, 2, 3, 4, 5, 6, 7,
                                         8, 9, 10, 11, 12, 13, 14, 15);
}

// ---------------------------------------------------------------------------
__global__ __launch_bounds__(256) void rnn_init_kernel(unsigned int* ws_u32) {
    // zero sync counter region + bf16 hidden state (h0 == 0)
    const int n = (HB_HI_OFF + 2 * BATCH * HDIM * 2) / 4;
    for (int i = blockIdx.x * blockDim.x + threadIdx.x; i < n;
         i += gridDim.x * blockDim.x)
        ws_u32[i] = 0u;
}

// ---------------------------------------------------------------------------
// xin[m][n] = sum_k x[m][k] * W_in[n][k];  M=T*B=32768, N=512, K=128.
// Written into the rnn_out section of d_out (used as a temp, overwritten by
// the scan with h_t at the same (t,b,h) location).
__global__ __launch_bounds__(256) void rnn_xin_kernel(
    const float* __restrict__ x,     // (T*B, I)
    const float* __restrict__ W_in,  // (H, I)
    float* __restrict__ xin)         // (T*B, H)
{
    const int tid  = threadIdx.x;
    const int lane = tid & 31;
    const int wave = tid >> 5;
    const int gw   = blockIdx.x * 8 + wave;     // 0..65535 tiles
    const int m0   = (gw >> 5) * 16;
    const int n0   = (gw & 31) * 16;
    const int nn   = lane & 15;
    const int hsel = lane >> 4;

    const float* arow = x    + (size_t)(m0 + nn) * IDIM;
    const float* brow = W_in + (size_t)(n0 + nn) * IDIM;

    v8f acc0 = {};
    v8f acc1 = {};
    for (int kk = 0; kk < IDIM; kk += 8) {
        v2f a0 = *(const v2f*)(arow + kk + 0 + hsel * 2);
        v2f b0 = *(const v2f*)(brow + kk + 0 + hsel * 2);
        v2f a1 = *(const v2f*)(arow + kk + 4 + hsel * 2);
        v2f b1 = *(const v2f*)(brow + kk + 4 + hsel * 2);
        acc0 = wmma_f32(a0, b0, acc0);
        acc1 = wmma_f32(a1, b1, acc1);
    }
    #pragma unroll
    for (int r = 0; r < 8; ++r) {
        int m = m0 + r + 8 * hsel;
        xin[(size_t)m * HDIM + n0 + nn] = acc0[r] + acc1[r];
    }
}

// ---------------------------------------------------------------------------
__global__ __launch_bounds__(128) void rnn_scan_kernel(
    const float* __restrict__ noise,  // (T,B,H)
    const float* __restrict__ W_rec,  // (H,H)
    const float* __restrict__ b_rec,  // (H)
    float* __restrict__ rnn_out,      // (T,B,H): holds xin on entry, h_t on exit
    bf16_t* __restrict__ hbHi,        // (B,H) bf16 state, hi part
    bf16_t* __restrict__ hbLo,        // (B,H) bf16 state, lo part
    unsigned int* __restrict__ sync_cnt)
{
    // W_rec^T slice in bf16 hi/lo, row-major [c][k] -> B fragment is two 8-value
    // K runs per lane (two 16B LDS reads).
    __shared__ bf16_t WbHi[CW * HDIM];   // 16 KB
    __shared__ bf16_t WbLo[CW * HDIM];   // 16 KB

    const int c0   = blockIdx.x * CW;
    const int tid  = threadIdx.x;
    const int lane = tid & 31;
    const int wave = tid >> 5;            // 0..3 -> batch tile
    const int m0   = wave * 16;
    const int nn   = lane & 15;
    const int hsel = lane >> 4;

    // one-time split of W_rec slice into LDS (coalesced global reads)
    for (int idx = tid; idx < CW * HDIM; idx += 128) {
        int k = idx & (HDIM - 1);
        int c = idx >> 9;
        float w   = W_rec[(c0 + c) * HDIM + k];
        bf16_t hi = (bf16_t)w;
        bf16_t lo = (bf16_t)(w - (float)hi);
        WbHi[c * HDIM + k] = hi;
        WbLo[c * HDIM + k] = lo;
    }
    __syncthreads();

    const float brec = b_rec[c0 + nn];
    const int   col  = c0 + nn;
    v8f hprev = {};                       // f32 state carry in registers

    for (int t = 0; t < T_STEPS; ++t) {
        v8f accH  = {};                   // Ahi*Bhi
        v8f accM1 = {};                   // Ahi*Blo
        v8f accM2 = {};                   // Alo*Bhi

        const bf16_t* arH = hbHi + (size_t)(m0 + nn) * HDIM;
        const bf16_t* arL = hbLo + (size_t)(m0 + nn) * HDIM;
        const bf16_t* brH = WbHi + nn * HDIM;
        const bf16_t* brL = WbLo + nn * HDIM;

        for (int kk = 0; kk < HDIM; kk += 32) {
            // A fragment: K runs [kk+hsel*8, +8) and [kk+16+hsel*8, +8)
            v16bf aHi = cat16(*(const v8bf*)(arH + kk + hsel * 8),
                              *(const v8bf*)(arH + kk + 16 + hsel * 8));
            v16bf aLo = cat16(*(const v8bf*)(arL + kk + hsel * 8),
                              *(const v8bf*)(arL + kk + 16 + hsel * 8));
            // B fragment: K runs [kk+hsel*8, +8) and [kk+16+hsel*8, +8)
            v16bf bHi = cat16(*(const v8bf*)(brH + kk + hsel * 8),
                              *(const v8bf*)(brH + kk + 16 + hsel * 8));
            v16bf bLo = cat16(*(const v8bf*)(brL + kk + hsel * 8),
                              *(const v8bf*)(brL + kk + 16 + hsel * 8));

            accH  = wmma_bf16(aHi, bHi, accH);
            accM1 = wmma_bf16(aHi, bLo, accM1);
            accM2 = wmma_bf16(aLo, bHi, accM2);
        }

        // pointwise: xin (pre-staged in rnn_out) + bias, relu, integrate,
        // noise; write h_t over xin and update the bf16 state split
        #pragma unroll
        for (int r = 0; r < 8; ++r) {
            int b = m0 + r + 8 * hsel;    // C/D layout: M = r + 8*(lane>>4)
            size_t oidx = ((size_t)t * BATCH + b) * HDIM + col;
            float xinv = rnn_out[oidx];
            float pre  = accH[r] + (accM1[r] + accM2[r]) + xinv + brec;
            pre        = fmaxf(pre, 0.0f);
            float nz   = noise[oidx];
            float hnew = hprev[r] * (1.0f - ALPHA_F) + ALPHA_F * pre
                       + nz * NOISE_SCALE_F;
            hprev[r] = hnew;
            rnn_out[oidx] = hnew;
            bf16_t hh = (bf16_t)hnew;
            bf16_t hl = (bf16_t)(hnew - (float)hh);
            hbHi[(size_t)b * HDIM + col] = hh;
            hbLo[(size_t)b * HDIM + col] = hl;
        }

        // grid-wide step barrier (release add / acquire spin)
        __threadfence();
        __syncthreads();
        if (tid == 0) {
            __hip_atomic_fetch_add(sync_cnt, 1u, __ATOMIC_RELEASE,
                                   __HIP_MEMORY_SCOPE_AGENT);
            const unsigned int target = (unsigned int)(t + 1) * NWG;
            while (__hip_atomic_load(sync_cnt, __ATOMIC_ACQUIRE,
                                     __HIP_MEMORY_SCOPE_AGENT) < target) {
                __builtin_amdgcn_s_sleep(2);
            }
        }
        __syncthreads();
        __threadfence();   // acquire for all waves (invalidates WGP$)
    }
}

// ---------------------------------------------------------------------------
// out[m][n] = sum_k rnn[m][k] * W_out[n][k] + b_out[n]; M=32768, N=64, K=512.
__global__ __launch_bounds__(256) void rnn_out_kernel(
    const float* __restrict__ rnn,    // (T*B, H)
    const float* __restrict__ W_out,  // (O, H)
    const float* __restrict__ b_out,  // (O)
    float* __restrict__ out)          // (T*B, O)
{
    const int tid  = threadIdx.x;
    const int lane = tid & 31;
    const int wave = tid >> 5;
    const int gw   = blockIdx.x * 8 + wave;    // 0..8191 tiles
    const int m0   = (gw >> 2) * 16;
    const int n0   = (gw & 3) * 16;
    const int nn   = lane & 15;
    const int hsel = lane >> 4;

    const float* arow = rnn   + (size_t)(m0 + nn) * HDIM;
    const float* brow = W_out + (size_t)(n0 + nn) * HDIM;

    v8f acc0 = {};
    v8f acc1 = {};
    for (int kk = 0; kk < HDIM; kk += 8) {
        v2f a0 = *(const v2f*)(arow + kk + 0 + hsel * 2);
        v2f b0 = *(const v2f*)(brow + kk + 0 + hsel * 2);
        v2f a1 = *(const v2f*)(arow + kk + 4 + hsel * 2);
        v2f b1 = *(const v2f*)(brow + kk + 4 + hsel * 2);
        acc0 = wmma_f32(a0, b0, acc0);
        acc1 = wmma_f32(a1, b1, acc1);
    }

    const float bo = b_out[n0 + nn];
    #pragma unroll
    for (int r = 0; r < 8; ++r) {
        int m = m0 + r + 8 * hsel;
        out[(size_t)m * ODIM + n0 + nn] = acc0[r] + acc1[r] + bo;
    }
}

// ---------------------------------------------------------------------------
extern "C" void kernel_launch(void* const* d_in, const int* in_sizes, int n_in,
                              void* d_out, int out_size, void* d_ws, size_t ws_size,
                              hipStream_t stream) {
    (void)in_sizes; (void)n_in; (void)out_size; (void)ws_size;

    const float* x     = (const float*)d_in[0];  // (T,B,I)
    const float* noise = (const float*)d_in[1];  // (T,B,H)
    const float* W_in  = (const float*)d_in[2];  // (H,I)
    const float* W_rec = (const float*)d_in[3];  // (H,H)
    const float* b_rec = (const float*)d_in[4];  // (H)
    const float* W_out = (const float*)d_in[5];  // (O,H)
    const float* b_out = (const float*)d_in[6];  // (O)

    float* out_proj = (float*)d_out;                                   // (T,B,O)
    float* rnn_out  = (float*)d_out + (size_t)T_STEPS * BATCH * ODIM;  // (T,B,H)

    unsigned int* sync_cnt = (unsigned int*)d_ws;
    bf16_t* hbHi = (bf16_t*)((char*)d_ws + HB_HI_OFF);
    bf16_t* hbLo = (bf16_t*)((char*)d_ws + HB_LO_OFF);

    // 1) zero sync counter + bf16 hidden state (deterministic per launch)
    rnn_init_kernel<<<64, 256, 0, stream>>>((unsigned int*)d_ws);

    // 2) input projection (parallel f32 WMMA) into rnn_out temp region
    rnn_xin_kernel<<<8192, 256, 0, stream>>>(x, W_in, rnn_out);

    // 3) persistent scan: 32 WGs x 4 waves; bf16x3 recurrent GEMM
    rnn_scan_kernel<<<NWG, 128, 0, stream>>>(noise, W_rec, b_rec, rnn_out,
                                             hbHi, hbLo, sync_cnt);

    // 4) output projection (parallel f32 WMMA)
    rnn_out_kernel<<<1024, 256, 0, stream>>>(rnn_out, W_out, b_out, out_proj);
}
